// GbldEmbLoss_25211458027845
// MI455X (gfx1250) — matrix operand order repaired and checked by reading, not compiled, not run on hardware
//
#include <hip/hip_runtime.h>
#include <hip/hip_bf16.h>

typedef __attribute__((ext_vector_type(16))) _Float16 v16h;
typedef __attribute__((ext_vector_type(8)))  float    v8f;

#define BATCH 4
#define PIX   6144            // H*W = 64*96
#define PMAX  6144            // padded capacity (multiple of 16)
#define NI    (PMAX / 16)     // 384 tile-rows
#define WAVES_PER_BLOCK 8
#define PULL_MARGIN 0.5f
#define PUSH_MARGIN 1.0f

// ---------------------------------------------------------------------------
// Kernel 1: deterministic stream compaction of foreground pixels per sample.
// One block (256 threads) per sample; Hillis-Steele scan keeps original order.
// ---------------------------------------------------------------------------
__global__ __launch_bounds__(256)
void gbld_compact(const float* __restrict__ pred,
                  const int*   __restrict__ gidx,
                  const int*   __restrict__ fmask,
                  float* __restrict__ vals,
                  int*   __restrict__ labels,
                  int*   __restrict__ counts) {
    const int b = blockIdx.x;
    const int t = threadIdx.x;
    __shared__ int s[256];
    __shared__ int base_s;
    if (t == 0) base_s = 0;
    __syncthreads();

    for (int c = 0; c < PIX; c += 256) {
        const int idx = c + t;
        const int m = (fmask[b * PIX + idx] != 0) ? 1 : 0;
        const int base = base_s;
        s[t] = m;
        __syncthreads();
        for (int off = 1; off < 256; off <<= 1) {
            int add = (t >= off) ? s[t - off] : 0;
            __syncthreads();
            s[t] += add;
            __syncthreads();
        }
        const int incl = s[t];
        const int pos  = base + incl - m;
        if (m) {
            vals[b * PMAX + pos]   = pred[b * PIX + idx];
            labels[b * PMAX + pos] = gidx[b * PIX + idx];
        }
        __syncthreads();
        if (t == 0) base_s = base + s[255];
        __syncthreads();
    }

    const int cnt = base_s;
    for (int i = cnt + t; i < PMAX; i += 256) {
        vals[b * PMAX + i]   = 0.0f;
        labels[b * PMAX + i] = -1;
    }
    if (t == 0) counts[b] = cnt;
}

// ---------------------------------------------------------------------------
// Kernel 2: pairwise tile loss.
// Block = 256 threads (8 waves). The block stages the whole per-sample panel
// (vals + labels, 48 KB) into LDS via CDNA5 async loads
// (global_load_async_to_lds_b32 + s_wait_asynccnt), then each wave32 handles
// one 16-row block, looping over 16-col blocks entirely out of LDS.
// Per tile: one v_wmma_f32_16x16x32_f16 on one-hot f16 label vectors gives
//   r = [l_i==l_j] + w_i*w_j  in {0,1,2} (exact); masks derived in pure f32.
// ---------------------------------------------------------------------------
__global__ __launch_bounds__(256)
void gbld_pairs(const float* __restrict__ vals,
                const int*   __restrict__ labels,
                const int*   __restrict__ counts,
                float* __restrict__ ppull,
                float* __restrict__ ppush,
                float* __restrict__ pnsame) {
    __shared__ float sv[PMAX];
    __shared__ int   sl[PMAX];

    const int t    = threadIdx.x;
    const int b    = blockIdx.y;
    const int wave = t >> 5;
    const int lane = t & 31;
    const int cnt  = counts[b];

    // ---- async stage: global -> LDS (per-lane 4B transfers, ASYNCcnt) ----
    {
        const float* gv = vals   + b * PMAX;
        const int*   gl = labels + b * PMAX;
        for (int p = t; p < PMAX; p += 256) {
            unsigned int lv = (unsigned int)(unsigned long long)(unsigned long long)(uintptr_t)&sv[p];
            unsigned int ll = (unsigned int)(uintptr_t)&sl[p];
            unsigned long long av = (unsigned long long)(uintptr_t)(gv + p);
            unsigned long long al = (unsigned long long)(uintptr_t)(gl + p);
            asm volatile("global_load_async_to_lds_b32 %0, %1, off"
                         :: "v"(lv), "v"(av) : "memory");
            asm volatile("global_load_async_to_lds_b32 %0, %1, off"
                         :: "v"(ll), "v"(al) : "memory");
        }
        asm volatile("s_wait_asynccnt 0" ::: "memory");
    }
    __syncthreads();

    const int ib     = blockIdx.x * WAVES_PER_BLOCK + wave;
    const int i0     = ib * 16;
    const int outIdx = b * NI + ib;

    if (i0 >= cnt) {
        if (lane == 0) { ppull[outIdx] = 0.0f; ppush[outIdx] = 0.0f; pnsame[outIdx] = 0.0f; }
        return;
    }

    const int half = lane >> 4;   // 0: lanes 0-15, 1: lanes 16-31
    const int l15  = lane & 15;

    // --- A matrix (16x32 f16, one-hot rows for this i-block) ---
    // lane L holds row M = L&15; slot s maps to K = s + 8*(s>=8) + 8*(L>=16)
    const int labA = sl[i0 + l15];
    v16h a;
#pragma unroll
    for (int s = 0; s < 16; ++s) {
        const int K = s + ((s >= 8) ? 8 : 0) + 8 * half;
        a[s] = (_Float16)(((K == labA) || (labA >= 0 && K == 20)) ? 1.0f : 0.0f);
    }

    // row values for distance: VGPR k of C/D holds row m = k + 8*half
    float vr[8];
#pragma unroll
    for (int k = 0; k < 8; ++k) vr[k] = sv[i0 + k + 8 * half];

    float pullA = 0.0f, pushA = 0.0f, nsameA = 0.0f;

    for (int j0 = 0; j0 < cnt; j0 += 16) {
        // --- B matrix (32x16 f16, one-hot columns for this j-block) ---
        // lane L holds column N = L&15; slot s maps to K = s + 8*(s>=8) + 8*(L>=16)
        const int   labB = sl[j0 + l15];
        const float vc   = sv[j0 + l15];
        v16h bm;
#pragma unroll
        for (int s = 0; s < 16; ++s) {
            const int K = s + ((s >= 8) ? 8 : 0) + 8 * half;
            bm[s] = (_Float16)(((K == labB) || (labB >= 0 && K == 20)) ? 1.0f : 0.0f);
        }

        v8f c = {};
        v8f d = __builtin_amdgcn_wmma_f32_16x16x32_f16(
            /*neg_a=*/false, a, /*neg_b=*/false, bm,
            /*c_mod=*/(short)0, c, /*reuse_a=*/false, /*reuse_b=*/false);

#pragma unroll
        for (int k = 0; k < 8; ++k) {
            const float r    = d[k];                    // exact 0,1,2
            const float same = fmaxf(r - 1.0f, 0.0f);   // same-label & both valid
            const float prm  = fminf(r, 1.0f);          // pair mask w_i*w_j
            const float diff = prm - same;              // diff-label & both valid
            const float dist = fabsf(vr[k] - vc);
            pullA += same * fmaxf(dist - PULL_MARGIN, 0.0f);
            pushA += diff * fmaxf(PUSH_MARGIN - dist, 0.0f);
            nsameA += same;                             // <= 98304, exact in f32
        }
    }

    // deterministic in-wave reduction (wave32)
#pragma unroll
    for (int off = 16; off >= 1; off >>= 1) {
        pullA  += __shfl_down(pullA,  off, 32);
        pushA  += __shfl_down(pushA,  off, 32);
        nsameA += __shfl_down(nsameA, off, 32);
    }
    if (lane == 0) {
        ppull[outIdx]  = pullA;
        ppush[outIdx]  = pushA;
        pnsame[outIdx] = nsameA;
    }
}

// ---------------------------------------------------------------------------
// Kernel 3: deterministic serial finalize (tiny: 4 samples x 384 partials).
// ---------------------------------------------------------------------------
__global__ __launch_bounds__(32)
void gbld_finalize(const float* __restrict__ ppull,
                   const float* __restrict__ ppush,
                   const float* __restrict__ pnsame,
                   const int*   __restrict__ counts,
                   float* __restrict__ out) {
    if (threadIdx.x != 0 || blockIdx.x != 0) return;
    float total = 0.0f;
    for (int b = 0; b < BATCH; ++b) {
        float ps = 0.0f, qs = 0.0f;
        long long ns = 0;
        for (int i = 0; i < NI; ++i) {
            ps += ppull[b * NI + i];
            qs += ppush[b * NI + i];
            ns += (long long)(pnsame[b * NI + i] + 0.5f);
        }
        const long long F  = (long long)counts[b];
        const long long nd = F * F - ns;
        const float pull = ps / fmaxf((float)ns, 1.0f);
        const float push = (nd > 0) ? (qs / (float)nd) : 0.0f;
        total += pull + push;
    }
    out[0] = total / (float)BATCH;  // LOSS_WEIGHT = 1.0
}

// ---------------------------------------------------------------------------
extern "C" void kernel_launch(void* const* d_in, const int* in_sizes, int n_in,
                              void* d_out, int out_size, void* d_ws, size_t ws_size,
                              hipStream_t stream) {
    const float* pred  = (const float*)d_in[0];   // [4,1,64,96] f32
    const int*   gidx  = (const int*)d_in[1];     // [4,64,96]   i32
    const int*   fmask = (const int*)d_in[2];     // [4,64,96]   bool->i32
    float* out = (float*)d_out;

    // workspace layout
    float* vals   = (float*)d_ws;                    // BATCH*PMAX
    int*   labels = (int*)(vals + BATCH * PMAX);     // BATCH*PMAX
    float* ppull  = (float*)(labels + BATCH * PMAX); // BATCH*NI
    float* ppush  = ppull + BATCH * NI;              // BATCH*NI
    float* pnsame = ppush + BATCH * NI;              // BATCH*NI
    int*   counts = (int*)(pnsame + BATCH * NI);     // BATCH

    gbld_compact<<<BATCH, 256, 0, stream>>>(pred, gidx, fmask, vals, labels, counts);
    gbld_pairs<<<dim3(NI / WAVES_PER_BLOCK, BATCH), 256, 0, stream>>>(
        vals, labels, counts, ppull, ppush, pnsame);
    gbld_finalize<<<1, 32, 0, stream>>>(ppull, ppush, pnsame, counts, out);
}
